// custom_rnn_40999757807990
// MI455X (gfx1250) — compile-verified, matrix operand order checked by reading, not run on hardware
//
#include <hip/hip_runtime.h>
#include <math.h>

// ---------------- problem dims ----------------
#define T_DIM 8192
#define I_DIM 256
#define H_DIM 2048
#define O_DIM 256
#define G3    (3 * H_DIM)   // 6144

// recurrence launch geometry: weights held register-resident.
// 128 WGs x 512 thr (16 waves); 1 wave per hidden unit; 64 K-elems/thread.
#define NWG          128
#define R_THREADS    512
#define UNITS_PER_WG (H_DIM / NWG)          // 16
#define SEG          32                      // threads per hidden unit (full wave)
#define KSEG         (H_DIM / SEG)           // 64 K elems per thread
#define KV           (KSEG / 8)              // 8 v8bf chunks per gate

#define THREADS      256                     // block size for GEMM phases

typedef __bf16 v16bf __attribute__((ext_vector_type(16)));
typedef __bf16 v8bf  __attribute__((ext_vector_type(8)));
typedef float  v8f   __attribute__((ext_vector_type(8)));

// ---------------- fp32 -> bf16 conversion ----------------
__global__ void k_cvt_bf16(const float* __restrict__ src, __bf16* __restrict__ dst, int n) {
    int i = blockIdx.x * blockDim.x + threadIdx.x;
    if (i < n) dst[i] = (__bf16)src[i];
}

// W_out [H, O] row-major -> W_outT bf16 [O, H] row-major
__global__ void k_cvt_wout_t(const float* __restrict__ wout, __bf16* __restrict__ woutT) {
    int i = blockIdx.x * blockDim.x + threadIdx.x;   // i over H*O
    int k = i / O_DIM;          // 0..H-1
    int n = i % O_DIM;          // 0..O-1
    woutT[(size_t)n * H_DIM + k] = (__bf16)wout[(size_t)k * O_DIM + n];
}

// ---------------- WMMA fragment loaders (wave32 layouts) ----------------
// A/B 16x32 bf16 fragment: lane L holds row m=L%16 (A) / col n=L%16 (B),
// K chunks at k0 + half*8 and k0 + 16 + half*8 (half = L/16).
__device__ __forceinline__ v16bf load_frag_bf16(const __bf16* __restrict__ base,
                                                int row0, int ld, int k0, int lane) {
    int m = lane & 15, half = lane >> 4;
    const __bf16* p = base + (size_t)(row0 + m) * ld + k0 + half * 8;
    union { v16bf v; v8bf h[2]; } u;
    u.h[0] = *(const v8bf*)(p);
    u.h[1] = *(const v8bf*)(p + 16);
    return u.v;
}

// same fragment but converting from f32 source on the fly
__device__ __forceinline__ v16bf load_frag_f32(const float* __restrict__ base,
                                               int row0, int ld, int k0, int lane) {
    int m = lane & 15, half = lane >> 4;
    const float* p = base + (size_t)(row0 + m) * ld + k0 + half * 8;
    v16bf v;
#pragma unroll
    for (int e = 0; e < 8; ++e) v[e] = (__bf16)p[e];
#pragma unroll
    for (int e = 0; e < 8; ++e) v[8 + e] = (__bf16)p[16 + e];
    return v;
}

// ---------------- phase 1: gi_all = X @ W_ih^T + b_ih ----------------
// grid (G3/32, T/128), block 256 (8 waves). Each wave: 16(M) x 32(N) tile.
__global__ void k_gemm_gi(const __bf16* __restrict__ Xb,     // [T, I]
                          const __bf16* __restrict__ Wihb,   // [G3, I]
                          const float*  __restrict__ b_ih,   // [G3]
                          float* __restrict__ gi) {          // [T, G3]
    int lane = threadIdx.x & 31;
    int wave = threadIdx.x >> 5;
    int m0 = blockIdx.y * 128 + wave * 16;
    int n0 = blockIdx.x * 32;

    v8f acc0 = {}; v8f acc1 = {};
#pragma unroll
    for (int k0 = 0; k0 < I_DIM; k0 += 32) {
        v16bf a  = load_frag_bf16(Xb,   m0,      I_DIM, k0, lane);
        v16bf b0 = load_frag_bf16(Wihb, n0,      I_DIM, k0, lane);
        v16bf b1 = load_frag_bf16(Wihb, n0 + 16, I_DIM, k0, lane);
        acc0 = __builtin_amdgcn_wmma_f32_16x16x32_bf16(false, a, false, b0,
                                                       (short)0, acc0, false, false);
        acc1 = __builtin_amdgcn_wmma_f32_16x16x32_bf16(false, a, false, b1,
                                                       (short)0, acc1, false, false);
    }
    int half = lane >> 4, nn = lane & 15;
#pragma unroll
    for (int r = 0; r < 8; ++r) {
        int m = m0 + half * 8 + r;
        int n = n0 + nn;
        gi[(size_t)m * G3 + n]      = acc0[r] + b_ih[n];
        gi[(size_t)m * G3 + n + 16] = acc1[r] + b_ih[n + 16];
    }
}

// ---------------- phase 2: persistent GRU recurrence ----------------
// W_hh is loaded ONCE into VGPRs (96 regs/thread of bf16 weights, one wave
// per hidden unit) and reused across all 8192 steps: per-step memory traffic
// is just the h broadcast + gi reads + h write.
__global__ void __launch_bounds__(R_THREADS, 1)
k_gru_recur(const float* __restrict__ gi,     // [T, G3]
            const __bf16* __restrict__ Whhb,  // [G3, H]
            const float* __restrict__ b_hh,   // [G3]
            const float* __restrict__ h0,     // [H]
            float* __restrict__ act,          // [T, H]
            float* __restrict__ hfinal,       // [H]
            unsigned* __restrict__ barrier_ct) {
    __shared__ float h_s[H_DIM];
    const int tid  = threadIdx.x;
    const int unit = tid / SEG;                 // 0..15 (== wave id)
    const int seg  = tid % SEG;                 // 0..31 (== lane)
    const int j    = blockIdx.x * UNITS_PER_WG + unit;   // hidden index

    // ---- preload this thread's recurrent weights into registers ----
    const v8bf* wr = (const v8bf*)(Whhb + (size_t)(j)             * H_DIM + seg * KSEG);
    const v8bf* wz = (const v8bf*)(Whhb + (size_t)(j + H_DIM)     * H_DIM + seg * KSEG);
    const v8bf* wn = (const v8bf*)(Whhb + (size_t)(j + 2 * H_DIM) * H_DIM + seg * KSEG);
    v8bf Wr[KV], Wz[KV], Wn[KV];
#pragma unroll
    for (int c = 0; c < KV; ++c) { Wr[c] = wr[c]; Wz[c] = wz[c]; Wn[c] = wn[c]; }

    const float bR = b_hh[j];
    const float bZ = b_hh[j + H_DIM];
    const float bN = b_hh[j + 2 * H_DIM];

    for (int t = 0; t < T_DIM; ++t) {
        // stage h_{t-1} into LDS (t==0 reads the initial hidden state input)
        const float* hsrc = (t == 0) ? h0 : (act + (size_t)(t - 1) * H_DIM);
        for (int k = tid; k < H_DIM; k += R_THREADS) h_s[k] = hsrc[k];
        __syncthreads();

        // three dot products over this thread's K segment, weights in VGPRs
        float aR = 0.f, aZ = 0.f, aN = 0.f;
        const float* hp = &h_s[seg * KSEG];
#pragma unroll
        for (int c = 0; c < KV; ++c) {
            v8bf r8 = Wr[c], z8 = Wz[c], n8 = Wn[c];
#pragma unroll
            for (int e = 0; e < 8; ++e) {
                float hv = hp[c * 8 + e];
                aR = fmaf((float)r8[e], hv, aR);
                aZ = fmaf((float)z8[e], hv, aZ);
                aN = fmaf((float)n8[e], hv, aN);
            }
        }
        // full-wave reduction across the 32 lanes of this unit
#pragma unroll
        for (int m = 1; m < SEG; m <<= 1) {
            aR += __shfl_xor(aR, m, 32);
            aZ += __shfl_xor(aZ, m, 32);
            aN += __shfl_xor(aN, m, 32);
        }

        if (seg == 0) {
            const float* git = gi + (size_t)t * G3;
            float r = 1.f / (1.f + expf(-(git[j]             + aR + bR)));
            float z = 1.f / (1.f + expf(-(git[j + H_DIM]     + aZ + bZ)));
            float n = tanhf(git[j + 2 * H_DIM] + r * (aN + bN));
            float hn = (1.f - z) * n + z * h_s[j];
            act[(size_t)t * H_DIM + j] = hn;
            if (t == T_DIM - 1) hfinal[j] = hn;
        }

        // grid-wide barrier (monotonic counter, reset by memset each call)
        __syncthreads();
        if (tid == 0) {
            __threadfence();
            __hip_atomic_fetch_add(barrier_ct, 1u, __ATOMIC_RELEASE,
                                   __HIP_MEMORY_SCOPE_AGENT);
            unsigned target = (unsigned)NWG * (unsigned)(t + 1);
            while (__hip_atomic_load(barrier_ct, __ATOMIC_ACQUIRE,
                                     __HIP_MEMORY_SCOPE_AGENT) < target) {
                __builtin_amdgcn_s_sleep(1);
            }
        }
        __syncthreads();
        __threadfence();
    }
}

// ---------------- phase 3: outputs = activity @ W_out + b_out ----------------
// grid (O/32, T/128), block 256 (8 waves). Each wave: 16(M) x 32(N) tile, K=H.
__global__ void k_gemm_out(const float* __restrict__ act,      // [T, H] f32
                           const __bf16* __restrict__ WoutT,   // [O, H] bf16
                           const float* __restrict__ b_out,    // [O]
                           float* __restrict__ out) {          // [T, O]
    int lane = threadIdx.x & 31;
    int wave = threadIdx.x >> 5;
    int m0 = blockIdx.y * 128 + wave * 16;
    int n0 = blockIdx.x * 32;

    v8f acc0 = {}; v8f acc1 = {};
    for (int k0 = 0; k0 < H_DIM; k0 += 32) {
        v16bf a  = load_frag_f32 (act,   m0,      H_DIM, k0, lane);
        v16bf b0 = load_frag_bf16(WoutT, n0,      H_DIM, k0, lane);
        v16bf b1 = load_frag_bf16(WoutT, n0 + 16, H_DIM, k0, lane);
        acc0 = __builtin_amdgcn_wmma_f32_16x16x32_bf16(false, a, false, b0,
                                                       (short)0, acc0, false, false);
        acc1 = __builtin_amdgcn_wmma_f32_16x16x32_bf16(false, a, false, b1,
                                                       (short)0, acc1, false, false);
    }
    int half = lane >> 4, nn = lane & 15;
#pragma unroll
    for (int r = 0; r < 8; ++r) {
        int m = m0 + half * 8 + r;
        int n = n0 + nn;
        out[(size_t)m * O_DIM + n]      = acc0[r] + b_out[n];
        out[(size_t)m * O_DIM + n + 16] = acc1[r] + b_out[n + 16];
    }
}

// ---------------- launcher ----------------
extern "C" void kernel_launch(void* const* d_in, const int* in_sizes, int n_in,
                              void* d_out, int out_size, void* d_ws, size_t ws_size,
                              hipStream_t stream) {
    const float* X     = (const float*)d_in[0];   // [T, I]
    const float* h0    = (const float*)d_in[1];   // [H]
    const float* W_ih  = (const float*)d_in[2];   // [G3, I]
    const float* W_hh  = (const float*)d_in[3];   // [G3, H]
    const float* b_ih  = (const float*)d_in[4];   // [G3]
    const float* b_hh  = (const float*)d_in[5];   // [G3]
    const float* W_out = (const float*)d_in[6];   // [H, O]
    const float* b_out = (const float*)d_in[7];   // [O]

    // output layout: outputs [T,O] | activity [T,H] | h_final [H]
    float* out_p  = (float*)d_out;
    float* act_p  = out_p + (size_t)T_DIM * O_DIM;
    float* hfin_p = act_p + (size_t)T_DIM * H_DIM;

    // workspace layout
    char* ws = (char*)d_ws;
    float*  gi    = (float*)ws;                                   ws += (size_t)T_DIM * G3 * sizeof(float);
    __bf16* Whhb  = (__bf16*)ws;                                  ws += (size_t)G3 * H_DIM * sizeof(__bf16);
    __bf16* Xb    = (__bf16*)ws;                                  ws += (size_t)T_DIM * I_DIM * sizeof(__bf16);
    __bf16* Wihb  = (__bf16*)ws;                                  ws += (size_t)G3 * I_DIM * sizeof(__bf16);
    __bf16* WoutT = (__bf16*)ws;                                  ws += (size_t)O_DIM * H_DIM * sizeof(__bf16);
    unsigned* barrier_ct = (unsigned*)ws;

    // phase 0: precision conversion (one-shot, bandwidth-trivial vs. phase 2)
    {
        int n;
        n = T_DIM * I_DIM;  k_cvt_bf16<<<(n + 255) / 256, 256, 0, stream>>>(X,    Xb,   n);
        n = G3 * I_DIM;     k_cvt_bf16<<<(n + 255) / 256, 256, 0, stream>>>(W_ih, Wihb, n);
        n = G3 * H_DIM;     k_cvt_bf16<<<(n + 255) / 256, 256, 0, stream>>>(W_hh, Whhb, n);
        n = H_DIM * O_DIM;  k_cvt_wout_t<<<(n + 255) / 256, 256, 0, stream>>>(W_out, WoutT);
    }

    // phase 1: big parallel WMMA GEMM for the input projection
    k_gemm_gi<<<dim3(G3 / 32, T_DIM / 128), THREADS, 0, stream>>>(Xb, Wihb, b_ih, gi);

    // phase 2: persistent-grid sequential recurrence, W_hh register-resident
    hipMemsetAsync(barrier_ct, 0, sizeof(unsigned), stream);
    k_gru_recur<<<NWG, R_THREADS, 0, stream>>>(gi, Whhb, b_hh, h0, act_p, hfin_p, barrier_ct);

    // phase 3: hoisted output projection as one parallel WMMA GEMM
    k_gemm_out<<<dim3(O_DIM / 32, T_DIM / 128), THREADS, 0, stream>>>(act_p, WoutT, b_out, out_p);
}
